// BDH_7009386627351
// MI455X (gfx1250) — matrix operand order repaired and checked by reading, not compiled, not run on hardware
//
#include <hip/hip_runtime.h>
#include <hip/hip_bf16.h>
#include <math.h>

// ---------------- model constants (from reference) ----------------
#define D_EMB 256
#define NHEAD 4
#define NI    2048     // per-head internal dim
#define NLAY  6
#define NVOC  256
#define TSEQ  2048
#define EPSV  1e-5f
#define TWO_PI 6.28318530717958647692f

typedef __bf16 bf16_t;
typedef __attribute__((ext_vector_type(8)))  __bf16 v8bf;
typedef __attribute__((ext_vector_type(16))) __bf16 v16bf;
typedef __attribute__((ext_vector_type(8)))  float  v8f;

// ---------------- batched bf16 WMMA GEMM, async-LDS double-buffered ------
// C[b] = A[b] (MxK, row-major) x Bt[b]^T, where Bt is (N x K) row-major
// (the B operand is always supplied K-minor / N-major).
// f32 accumulate; epilogue: optional relu, strict-lower causal zeroing,
// optional f32 and/or bf16 stores.
// Block tile 128x256, 8 waves, wave tile 64x64 (4x4 WMMA fragments):
// 16 ds_load_b128 feed 16 v_wmma per K-step (0.5KB LDS traffic per WMMA).
#define BM 128
#define BN 256
#define BK 32
#define LDSK 40   // BK + 8 halves padding (keeps 16B alignment, spreads banks)

__global__ __launch_bounds__(256)
void gemm_bf16_wmma(const bf16_t* __restrict__ A, const bf16_t* __restrict__ Bt,
                    float* __restrict__ Cf, bf16_t* __restrict__ Cb,
                    int M, int Nc, int K, int lda, int ldb, int ldc,
                    long long sA, long long sB, long long sC,
                    int relu, int causal)
{
    __shared__ alignas(16) bf16_t As[2][BM][LDSK];   // 20 KB
    __shared__ alignas(16) bf16_t Bs[2][BN][LDSK];   // 40 KB, N-major: Bs[n][k]

    const int tid  = threadIdx.x;
    const int lane = tid & 31;
    const int wid  = tid >> 5;        // 8 waves
    const int wm   = wid >> 2;        // 0..1 -> 64 rows each
    const int wn   = wid & 3;         // 0..3 -> 64 cols each

    const int tile_m = blockIdx.y * BM;
    const int tile_n = blockIdx.x * BN;
    const int batch  = blockIdx.z;

    A  += (long long)batch * sA;
    Bt += (long long)batch * sB;
    const long long cbase = (long long)batch * sC;

    // Async copy of one 128x32 A tile + one 256x32 Bt tile into an LDS buffer.
    // A: 512 16B chunks (2/thread), Bt: 1024 16B chunks (4/thread) ->
    // 6 async instructions per wave per tile (tracked by ASYNCcnt, in-order).
    auto prefetch = [&](int kk, int buf) {
#pragma unroll
        for (int i = 0; i < 2; ++i) {
            int c   = tid + i * 256;
            int row = c >> 2;
            int off = (c & 3) << 3;
            unsigned la = (unsigned)(size_t)&As[buf][row][off];
            const bf16_t* ga = A + (long long)(tile_m + row) * lda + kk + off;
            asm volatile("global_load_async_to_lds_b128 %0, %1, off"
                         :: "v"(la), "v"(ga) : "memory");
        }
#pragma unroll
        for (int i = 0; i < 4; ++i) {
            int c   = tid + i * 256;
            int row = c >> 2;
            int off = (c & 3) << 3;
            unsigned lb = (unsigned)(size_t)&Bs[buf][row][off];
            const bf16_t* gb = Bt + (long long)(tile_n + row) * ldb + kk + off;
            asm volatile("global_load_async_to_lds_b128 %0, %1, off"
                         :: "v"(lb), "v"(gb) : "memory");
        }
    };

    v8f acc[4][4];
#pragma unroll
    for (int fm = 0; fm < 4; ++fm)
#pragma unroll
        for (int fn = 0; fn < 4; ++fn)
#pragma unroll
            for (int e = 0; e < 8; ++e) acc[fm][fn][e] = 0.0f;

    // One K-step of WMMAs from LDS buffer `buf`.
    auto compute = [&](int buf) {
        const int base8 = (lane >> 4) << 3;    // A: lanes16-31 own K+8 halves
        v16bf afrag[4];
#pragma unroll
        for (int fm = 0; fm < 4; ++fm) {
            int m = wm * 64 + fm * 16 + (lane & 15);
            v8bf lo = *(const v8bf*)&As[buf][m][base8];        // K = base8..+7
            v8bf hi = *(const v8bf*)&As[buf][m][base8 + 16];   // K = base8+16..+23
            afrag[fm] = __builtin_shufflevector(lo, hi,
                0,1,2,3,4,5,6,7,8,9,10,11,12,13,14,15);
        }
        const int bk0 = (lane >> 4) << 3;      // B: lanes16-31 own K+8 halves
        v16bf bfrag[4];
#pragma unroll
        for (int fn = 0; fn < 4; ++fn) {
            int n = wn * 64 + fn * 16 + (lane & 15);
            v8bf lo = *(const v8bf*)&Bs[buf][n][bk0];
            v8bf hi = *(const v8bf*)&Bs[buf][n][bk0 + 16];
            bfrag[fn] = __builtin_shufflevector(lo, hi,
                0,1,2,3,4,5,6,7,8,9,10,11,12,13,14,15);
        }
#pragma unroll
        for (int fm = 0; fm < 4; ++fm)
#pragma unroll
            for (int fn = 0; fn < 4; ++fn)
                acc[fm][fn] = __builtin_amdgcn_wmma_f32_16x16x32_bf16(
                    false, afrag[fm], false, bfrag[fn],
                    (short)0, acc[fm][fn], false, false);
    };

    const int ntiles = K / BK;
    prefetch(0, 0);

    // steady state: branch-free software pipeline, one tile always in flight
    for (int it = 0; it < ntiles - 1; ++it) {
        prefetch((it + 1) * BK, (it + 1) & 1);
        // previous tile's 6 async ops complete (in-order) once cnt <= 6
        asm volatile("s_wait_asynccnt 0x6" ::: "memory");
        __syncthreads();          // all waves' tile data visible in LDS
        compute(it & 1);
        __syncthreads();          // done reading before this buffer is refilled
    }
    // tail tile
    asm volatile("s_wait_asynccnt 0x0" ::: "memory");
    __syncthreads();
    compute((ntiles - 1) & 1);

    // ---- epilogue: C/D layout = VGPR e -> row e + 8*(lane/16), col lane%16 ----
#pragma unroll
    for (int fm = 0; fm < 4; ++fm) {
#pragma unroll
        for (int fn = 0; fn < 4; ++fn) {
            int col   = tile_n + wn * 64 + fn * 16 + (lane & 15);
            int rbase = tile_m + wm * 64 + fm * 16 + ((lane >> 4) << 3);
#pragma unroll
            for (int e = 0; e < 8; ++e) {
                int row = rbase + e;
                float v = acc[fm][fn][e];
                if (relu)   v = v > 0.0f ? v : 0.0f;
                if (causal) v = (col < row) ? v : 0.0f;   // strict lower tri
                long long o = cbase + (long long)row * ldc + col;
                if (Cf) Cf[o] = v;
                if (Cb) Cb[o] = (bf16_t)v;
            }
        }
    }
}

// ---------------- elementwise / reduction kernels ----------------
// out[b][c][r] = (bf16) in[b][r][c]   (batched transpose + downconvert)
__global__ void transpose_to_bf16_k(const float* __restrict__ in,
                                    bf16_t* __restrict__ out,
                                    int R, int C, long long total) {
    long long i = (long long)blockIdx.x * 256 + threadIdx.x;
    if (i >= total) return;
    long long rc = (long long)R * C;
    long long b  = i / rc;
    long long rm = i - b * rc;
    int r = (int)(rm / C);
    int c = (int)(rm % C);
    out[b * rc + (long long)c * R + r] = (bf16_t)in[i];
}

__device__ __forceinline__ float blk_sum(float v, float* red, int i) {
    red[i] = v; __syncthreads();
    for (int s = 128; s > 0; s >>= 1) { if (i < s) red[i] += red[i + s]; __syncthreads(); }
    float r = red[0]; __syncthreads();
    return r;
}

// x = LN(embed[idx]); writes x (f32), xb (bf16), xbt (bf16, DxT transposed)
__global__ __launch_bounds__(256)
void embed_ln_k(const int* __restrict__ idx, const float* __restrict__ embed,
                float* __restrict__ x, bf16_t* __restrict__ xb,
                bf16_t* __restrict__ xbt) {
    __shared__ float red[256];
    int t = blockIdx.x, i = threadIdx.x;
    float v = embed[(long long)idx[t] * D_EMB + i];
    float mean = blk_sum(v, red, i) * (1.0f / D_EMB);
    float d = v - mean;
    float var = blk_sum(d * d, red, i) * (1.0f / D_EMB);
    float o = d * rsqrtf(var + EPSV);
    x  [(long long)t * D_EMB + i] = o;
    xb [(long long)t * D_EMB + i] = (bf16_t)o;
    xbt[(long long)i * TSEQ  + t] = (bf16_t)o;
}

// in-place row LayerNorm over dim 256 + bf16 copy; one block per row
__global__ __launch_bounds__(256)
void ln_rows_k(float* __restrict__ data, bf16_t* __restrict__ outb) {
    __shared__ float red[256];
    long long r = blockIdx.x; int i = threadIdx.x;
    float v = data[r * D_EMB + i];
    float mean = blk_sum(v, red, i) * (1.0f / D_EMB);
    float d = v - mean;
    float var = blk_sum(d * d, red, i) * (1.0f / D_EMB);
    float o = d * rsqrtf(var + EPSV);
    data[r * D_EMB + i] = o;
    outb[r * D_EMB + i] = (bf16_t)o;
}

// x = LN(x + LN(yMLP)); writes x, xb, xbt
__global__ __launch_bounds__(256)
void combine_ln_k(const float* __restrict__ ymlp, float* __restrict__ x,
                  bf16_t* __restrict__ xb, bf16_t* __restrict__ xbt) {
    __shared__ float red[256];
    long long t = blockIdx.x; int i = threadIdx.x;
    float y = ymlp[t * D_EMB + i];
    float m1 = blk_sum(y, red, i) * (1.0f / D_EMB);
    float d1 = y - m1;
    float v1 = blk_sum(d1 * d1, red, i) * (1.0f / D_EMB);
    float z = x[t * D_EMB + i] + d1 * rsqrtf(v1 + EPSV);
    float m2 = blk_sum(z, red, i) * (1.0f / D_EMB);
    float d2 = z - m2;
    float v2 = blk_sum(d2 * d2, red, i) * (1.0f / D_EMB);
    float o = d2 * rsqrtf(v2 + EPSV);
    x  [t * D_EMB + i]           = o;
    xb [t * D_EMB + i]           = (bf16_t)o;
    xbt[(long long)i * TSEQ + t] = (bf16_t)o;
}

// qr = rope(x_sparse) -> bf16 ; one thread per even/odd pair
__global__ void rope_k(const float* __restrict__ xsp, bf16_t* __restrict__ qr,
                       long long npairs) {
    long long p = (long long)blockIdx.x * 256 + threadIdx.x;
    if (p >= npairs) return;
    long long base = p * 2;
    int n2 = (int)(base % NI);            // already even => floor(n/2)*2
    long long row = base / NI;            // h*T + t
    int t = (int)(row % TSEQ);
    float freq  = exp2f(-16.0f * (float)n2 / (float)NI) * (1.0f / TWO_PI);
    float ph    = fmodf((float)t * freq, 1.0f) * TWO_PI;
    float c = cosf(ph), s = sinf(ph);
    float a = xsp[base], b = xsp[base + 1];
    qr[base]     = (bf16_t)(a * c - b * s);
    qr[base + 1] = (bf16_t)(b * c + a * s);
}

// xy[t][h*NI+n] = x_sparse[h][t][n] * y_sparse[h][t][n]  (bf16)
__global__ void mul_xy_k(const float* __restrict__ xsp, const float* __restrict__ ysp,
                         bf16_t* __restrict__ xyb) {
    long long i = (long long)blockIdx.x * 256 + threadIdx.x;
    if (i >= (long long)NHEAD * TSEQ * NI) return;
    int n = (int)(i % NI);
    long long r = i / NI;
    int t = (int)(r % TSEQ);
    int h = (int)(r / TSEQ);
    xyb[(long long)t * (NHEAD * NI) + (long long)h * NI + n] = (bf16_t)(xsp[i] * ysp[i]);
}

// ---------------- host orchestration ----------------
extern "C" void kernel_launch(void* const* d_in, const int* in_sizes, int n_in,
                              void* d_out, int out_size, void* d_ws, size_t ws_size,
                              hipStream_t stream) {
    (void)in_sizes; (void)n_in; (void)out_size; (void)ws_size;
    const int*   idx   = (const int*)  d_in[0];
    const float* embed = (const float*)d_in[1];
    const float* enc   = (const float*)d_in[2];
    const float* encv  = (const float*)d_in[3];
    const float* dec   = (const float*)d_in[4];
    const float* lmh   = (const float*)d_in[5];
    float* logits = (float*)d_out;

    char* ws = (char*)d_ws;
    size_t off = 0;
    auto alloc = [&](size_t bytes) -> void* {
        void* p = ws + off;
        off += (bytes + 255) & ~(size_t)255;
        return p;
    };

    float*  x      = (float*) alloc((size_t)TSEQ * D_EMB * 4);
    bf16_t* xb     = (bf16_t*)alloc((size_t)TSEQ * D_EMB * 2);
    bf16_t* xbt    = (bf16_t*)alloc((size_t)D_EMB * TSEQ * 2);   // x^T
    bf16_t* enc_t  = (bf16_t*)alloc((size_t)NHEAD * NI * D_EMB * 2);  // per head (NI x D)
    bf16_t* encv_t = (bf16_t*)alloc((size_t)NHEAD * NI * D_EMB * 2);
    bf16_t* dec_t  = (bf16_t*)alloc((size_t)D_EMB * NHEAD * NI * 2);  // (D x 8192)
    bf16_t* lmh_t  = (bf16_t*)alloc((size_t)NVOC * D_EMB * 2);        // (VOC x D)
    float*  xsp    = (float*) alloc((size_t)NHEAD * TSEQ * NI * 4);
    bf16_t* qr_b   = (bf16_t*)alloc((size_t)NHEAD * TSEQ * NI * 2);
    bf16_t* sc_b   = (bf16_t*)alloc((size_t)NHEAD * TSEQ * TSEQ * 2);
    float*  ykv    = (float*) alloc((size_t)NHEAD * TSEQ * D_EMB * 4);
    bf16_t* ykv_b  = (bf16_t*)alloc((size_t)NHEAD * TSEQ * D_EMB * 2);
    float*  ysp    = (float*) alloc((size_t)NHEAD * TSEQ * NI * 4);
    bf16_t* xy_b   = (bf16_t*)alloc((size_t)TSEQ * NHEAD * NI * 2);
    float*  ymlp   = (float*) alloc((size_t)TSEQ * D_EMB * 4);

    // weight conversion f32 -> bf16 with transpose (all B operands N-major)
    long long nW = (long long)NHEAD * D_EMB * NI;
    transpose_to_bf16_k<<<dim3((unsigned)((nW + 255) / 256)), 256, 0, stream>>>(
        enc,  enc_t,  D_EMB, NI, nW);                  // (D x NI) -> (NI x D) per head
    transpose_to_bf16_k<<<dim3((unsigned)((nW + 255) / 256)), 256, 0, stream>>>(
        encv, encv_t, D_EMB, NI, nW);
    transpose_to_bf16_k<<<dim3((unsigned)((nW + 255) / 256)), 256, 0, stream>>>(
        dec,  dec_t,  NHEAD * NI, D_EMB, nW);          // (8192 x D) -> (D x 8192)
    long long nL = (long long)D_EMB * NVOC;
    transpose_to_bf16_k<<<dim3((unsigned)((nL + 255) / 256)), 256, 0, stream>>>(
        lmh, lmh_t, D_EMB, NVOC, nL);                  // (D x VOC) -> (VOC x D)

    embed_ln_k<<<TSEQ, 256, 0, stream>>>(idx, embed, x, xb, xbt);

    const long long sHN = (long long)TSEQ * NI;      // per-head x_sparse stride
    const long long sHW = (long long)NI * D_EMB;     // per-head weight stride
    const long long sHT = (long long)TSEQ * TSEQ;    // per-head scores stride
    const long long sHD = (long long)TSEQ * D_EMB;   // per-head yKV stride
    long long npairs = (long long)NHEAD * TSEQ * NI / 2;
    long long nxy    = (long long)NHEAD * TSEQ * NI;

    for (int l = 0; l < NLAY; ++l) {
        // x_sparse = relu(x @ encoder)          [per head: (T,256)x(256,2048)]
        gemm_bf16_wmma<<<dim3(NI / BN, TSEQ / BM, NHEAD), 256, 0, stream>>>(
            xb, enc_t, xsp, nullptr,
            TSEQ, NI, D_EMB, D_EMB, D_EMB, NI,
            0, sHW, sHN, /*relu=*/1, /*causal=*/0);

        // qr = rope(x_sparse) -> bf16
        rope_k<<<dim3((unsigned)((npairs + 255) / 256)), 256, 0, stream>>>(xsp, qr_b, npairs);

        // scores = tril_strict(qr @ qr^T) -> bf16  [per head: (T,2048)x(2048,T)]
        gemm_bf16_wmma<<<dim3(TSEQ / BN, TSEQ / BM, NHEAD), 256, 0, stream>>>(
            qr_b, qr_b, nullptr, sc_b,
            TSEQ, TSEQ, NI, NI, NI, TSEQ,
            sHN, sHN, sHT, /*relu=*/0, /*causal=*/1);

        // yKV_raw = scores @ x                  [per head: (T,T)x(T,256)]
        gemm_bf16_wmma<<<dim3(D_EMB / BN, TSEQ / BM, NHEAD), 256, 0, stream>>>(
            sc_b, xbt, ykv, nullptr,
            TSEQ, D_EMB, TSEQ, TSEQ, TSEQ, D_EMB,
            sHT, 0, sHD, /*relu=*/0, /*causal=*/0);

        // yKV = LN(yKV_raw) (in place) + bf16 copy
        ln_rows_k<<<NHEAD * TSEQ, 256, 0, stream>>>(ykv, ykv_b);

        // y_sparse = relu(yKV @ encoder_v)
        gemm_bf16_wmma<<<dim3(NI / BN, TSEQ / BM, NHEAD), 256, 0, stream>>>(
            ykv_b, encv_t, ysp, nullptr,
            TSEQ, NI, D_EMB, D_EMB, D_EMB, NI,
            sHD, sHW, sHN, /*relu=*/1, /*causal=*/0);

        // xy = x_sparse * y_sparse, relayout to (T, NH*NI), bf16
        mul_xy_k<<<dim3((unsigned)((nxy + 255) / 256)), 256, 0, stream>>>(xsp, ysp, xy_b);

        // yMLP = xy @ decoder                   [(T,8192)x(8192,256)]
        gemm_bf16_wmma<<<dim3(D_EMB / BN, TSEQ / BM, 1), 256, 0, stream>>>(
            xy_b, dec_t, ymlp, nullptr,
            TSEQ, D_EMB, NHEAD * NI, NHEAD * NI, NHEAD * NI, D_EMB,
            0, 0, 0, /*relu=*/0, /*causal=*/0);

        // x = LN(x + LN(yMLP))
        combine_ln_k<<<TSEQ, 256, 0, stream>>>(ymlp, x, xb, xbt);
    }

    // logits = hidden @ lm_head                 [(T,256)x(256,256)]
    gemm_bf16_wmma<<<dim3(NVOC / BN, TSEQ / BM, 1), 256, 0, stream>>>(
        xb, lmh_t, logits, nullptr,
        TSEQ, NVOC, D_EMB, D_EMB, D_EMB, NVOC,
        0, 0, 0, /*relu=*/0, /*causal=*/0);
}
